// SparseMoE_27650999451870
// MI455X (gfx1250) — matrix operand order repaired and checked by reading, not compiled, not run on hardware
//
#include <hip/hip_runtime.h>
#include <hip/hip_bf16.h>

// ---------------- problem constants ----------------
#define T_TOK 4096
#define H_DIM 1024
#define E_DIM 1024
#define NEXP  32
#define TOPK  4
#define CAP   512
#define NH    4
#define HD    256
#define NSTD  24   // experts with e%4 != 0
#define NLIN  8    // experts with e%4 == 0

typedef __attribute__((ext_vector_type(16))) __bf16          v16bf;
typedef __attribute__((ext_vector_type(8)))  float           v8f;
typedef __attribute__((ext_vector_type(8)))  unsigned short  u16x8;
typedef __attribute__((ext_vector_type(4)))  unsigned short  u16x4;

union FragBF { u16x8 h[2]; v16bf b; };

// native bf16 conversions (gfx1250 has hardware bf16; fptrunc is RNE)
static __device__ __forceinline__ unsigned short f2bf(float f) {
  __bf16 b = (__bf16)f;
  return __builtin_bit_cast(unsigned short, b);
}
static __device__ __forceinline__ float bf2f(unsigned short h) {
  return (float)__builtin_bit_cast(__bf16, h);
}

// ---------------- router: logits -> softmax -> top4 ----------------
__global__ __launch_bounds__(32) void router_kernel(
    const float* __restrict__ x, const float* __restrict__ rw,
    int* __restrict__ topi, float* __restrict__ topw)
{
  int t = blockIdx.x; int lane = threadIdx.x;           // lane == expert
  const float4* xr = (const float4*)(x + (long)t * H_DIM);
  const float4* wr = (const float4*)(rw + (long)lane * H_DIM);
  float acc = 0.f;
  for (int i = 0; i < H_DIM / 4; ++i) {
    float4 a = xr[i], b = wr[i];
    acc += a.x * b.x + a.y * b.y + a.z * b.z + a.w * b.w;
  }
  float m = acc;
  for (int off = 16; off; off >>= 1) m = fmaxf(m, __shfl_xor(m, off));
  float ex = __expf(acc - m);
  float s = ex;
  for (int off = 16; off; off >>= 1) s += __shfl_xor(s, off);
  float prob = ex / s;

  float p = prob;
  float wsel[TOPK]; int isel[TOPK];
  for (int k = 0; k < TOPK; ++k) {
    float v = p; int idx = lane;
    for (int off = 16; off; off >>= 1) {
      float ov = __shfl_xor(v, off); int oi = __shfl_xor(idx, off);
      if (ov > v || (ov == v && oi < idx)) { v = ov; idx = oi; }
    }
    wsel[k] = v; isel[k] = idx;
    if (lane == idx) p = -1.f;
  }
  float sw = wsel[0] + wsel[1] + wsel[2] + wsel[3];
  if (lane < TOPK) {
    topi[t * TOPK + lane] = isel[lane];
    topw[t * TOPK + lane] = wsel[lane] / (sw + 1e-6f);
  }
}

// ---------------- ordered capacity assignment ----------------
__global__ __launch_bounds__(128) void assign_kernel(
    const int* __restrict__ topi, int* __restrict__ gidx,
    int* __restrict__ tpos, int* __restrict__ counts)
{
  int kk = threadIdx.x >> 5;
  int e  = threadIdx.x & 31;
  int cnt = 0;
  for (int t = 0; t < T_TOK; ++t) {
    if (topi[t * TOPK + kk] == e) {
      if (cnt < CAP) {
        gidx[((long)(kk * NEXP + e)) * CAP + cnt] = t;
        tpos[t * TOPK + kk] = cnt;
      } else {
        tpos[t * TOPK + kk] = -1;
      }
      cnt++;
    }
  }
  counts[kk * NEXP + e] = cnt < CAP ? cnt : CAP;
}

// ---------------- gather tokens into expert buffers (fp32 -> bf16) ----------------
__global__ __launch_bounds__(256) void gather_kernel(
    const float* __restrict__ x, const int* __restrict__ gidx,
    const int* __restrict__ counts, unsigned short* __restrict__ Xb, int kk)
{
  int e = blockIdx.y, p = blockIdx.x;
  int c = counts[kk * NEXP + e];
  unsigned short* dst = Xb + ((long)e * CAP + p) * H_DIM;
  if (p < c) {
    int t = gidx[((long)(kk * NEXP + e)) * CAP + p];
    const float* src = x + (long)t * H_DIM;
    for (int j = threadIdx.x; j < H_DIM; j += 256) dst[j] = f2bf(src[j]);
  } else {
    for (int j = threadIdx.x; j < H_DIM; j += 256) dst[j] = 0;
  }
}

// ---------------- generic bf16 WMMA GEMM ----------------
// C[M,N] = epilogue( alpha * A op B ) ; A bf16, B bf16 or fp32(convert), C bf16/f32
// TA: A stored [K,M] (computes A^T @ B).  TB: B stored [N,K] (computes A @ B^T).
// mode 0: off = z*stride.  mode 1 (q@k^T per (expert,head)).  mode 2 (P@v per (expert,head)).
#define BM 128
#define BN 128
#define BK 32
#define LDSTR 40   // padded LDS row stride (ushorts), keeps 16B alignment

template<bool TA, bool TB, bool BF32, bool OUTF32, bool BIAS, bool RS>
__global__ __launch_bounds__(256) void gemm_wmma(
    const unsigned short* __restrict__ Abase, const void* __restrict__ Bbase,
    const float* __restrict__ biasBase, const float* __restrict__ rsBase,
    void* __restrict__ Cbase,
    int M, int N, int K,
    long lda, long ldb, long ldc,
    long sA, long sB, long sC, long sBias, long sRS,
    float alpha, int mode)
{
  __shared__ unsigned short As[BM * LDSTR];
  __shared__ unsigned short Bs[BN * LDSTR];   // stored [n][k]

  int z = blockIdx.z;
  long offA, offB, offC, offBias = 0, offRS = 0;
  if (mode == 0) {
    offA = (long)z * sA; offB = (long)z * sB; offC = (long)z * sC;
    offBias = (long)z * sBias; offRS = (long)z * sRS;
  } else {
    int i = z >> 2, h = z & 3;
    int e = (i / 3) * 4 + (i % 3) + 1;               // experts with e%4 != 0
    if (mode == 1) { offA = (long)e * sA + h * HD; offB = (long)e * sB + h * HD; offC = (long)z * sC; }
    else           { offA = (long)z * sA;          offB = (long)e * sB + h * HD; offC = (long)e * sC + h * HD; }
  }
  const unsigned short* A = Abase + offA;
  const float* bias = BIAS ? (biasBase + offBias) : nullptr;
  const float* rs   = RS   ? (rsBase + offRS)     : nullptr;

  int m0 = blockIdx.y * BM;
  int n0 = blockIdx.x * BN;
  int tid = threadIdx.x;
  int lane = tid & 31;
  int wid = tid >> 5;
  int wm = (wid & 3) * 32;   // 4 wave-rows of 32
  int wn = (wid >> 2) * 64;  // 2 wave-cols of 64

  v8f acc[2][4];
  #pragma unroll
  for (int a = 0; a < 2; ++a)
    #pragma unroll
    for (int b = 0; b < 4; ++b)
      #pragma unroll
      for (int r = 0; r < 8; ++r) acc[a][b][r] = 0.f;

  for (int k0 = 0; k0 < K; k0 += BK) {
    // ---- stage A tile ----
    if (!TA) {
      int r = tid >> 2;            // 0..63
      int c = (tid & 3) * 8;       // 0..24
      #pragma unroll
      for (int p = 0; p < 2; ++p) {
        u16x8 v = *(const u16x8*)(A + (long)(m0 + r + p * 64) * lda + k0 + c);
        *(u16x8*)(&As[(r + p * 64) * LDSTR + c]) = v;
      }
      if (k0 + BK < K)             // hint next A tile (global_prefetch_b8)
        __builtin_prefetch(A + (long)(m0 + r) * lda + k0 + BK + c, 0, 0);
    } else {                        // A stored [K,M]: transpose on LDS write
      int kk = tid >> 4;            // 0..15
      int mc = (tid & 15) * 8;
      #pragma unroll
      for (int p = 0; p < 2; ++p) {
        u16x8 v = *(const u16x8*)(A + (long)(k0 + kk + p * 16) * lda + m0 + mc);
        #pragma unroll
        for (int j = 0; j < 8; ++j) As[(mc + j) * LDSTR + kk + p * 16] = v[j];
      }
      if (k0 + BK < K)
        __builtin_prefetch(A + (long)(k0 + BK + kk) * lda + m0 + mc, 0, 0);
    }
    // ---- stage B tile into Bs[n][k] ----
    if (TB) {
      if (BF32) {                   // weights [N,K] fp32 -> bf16 (native cvt)
        const float* Bp = (const float*)Bbase + offB;
        int n = tid >> 3;           // 0..31
        int kc = (tid & 7) * 4;
        #pragma unroll
        for (int p = 0; p < 4; ++p) {
          float4 v = *(const float4*)(Bp + (long)(n0 + n + p * 32) * ldb + k0 + kc);
          u16x4 o; o[0] = f2bf(v.x); o[1] = f2bf(v.y); o[2] = f2bf(v.z); o[3] = f2bf(v.w);
          *(u16x4*)(&Bs[(n + p * 32) * LDSTR + kc]) = o;
        }
      } else {                      // activations [N,K] bf16
        const unsigned short* Bp = (const unsigned short*)Bbase + offB;
        int n = tid >> 2;           // 0..63
        int kc = (tid & 3) * 8;
        #pragma unroll
        for (int p = 0; p < 2; ++p) {
          u16x8 v = *(const u16x8*)(Bp + (long)(n0 + n + p * 64) * ldb + k0 + kc);
          *(u16x8*)(&Bs[(n + p * 64) * LDSTR + kc]) = v;
        }
      }
    } else {                        // B stored [K,N] bf16: transpose on LDS write
      const unsigned short* Bp = (const unsigned short*)Bbase + offB;
      int kk = tid >> 4;            // 0..15
      int nc = (tid & 15) * 8;
      #pragma unroll
      for (int p = 0; p < 2; ++p) {
        u16x8 v = *(const u16x8*)(Bp + (long)(k0 + kk + p * 16) * ldb + n0 + nc);
        #pragma unroll
        for (int j = 0; j < 8; ++j) Bs[(nc + j) * LDSTR + kk + p * 16] = v[j];
      }
      if (k0 + BK < K)
        __builtin_prefetch(Bp + (long)(k0 + BK + kk) * ldb + n0 + nc, 0, 0);
    }
    __syncthreads();

    // ---- WMMA fragments (CDNA5 16x16x32 bf16 layouts) ----
    int fr  = lane & 15;
    int kbA = (lane >> 4) << 3;     // A: halves hold K {0..7,16..23} / {8..15,24..31}
    int kbB = (lane >> 4) << 3;     // B: halves hold K {0..7,16..23} / {8..15,24..31}
    FragBF a0, a1;
    a0.h[0] = *(const u16x8*)(&As[(wm + fr) * LDSTR + kbA]);
    a0.h[1] = *(const u16x8*)(&As[(wm + fr) * LDSTR + kbA + 16]);
    a1.h[0] = *(const u16x8*)(&As[(wm + 16 + fr) * LDSTR + kbA]);
    a1.h[1] = *(const u16x8*)(&As[(wm + 16 + fr) * LDSTR + kbA + 16]);
    #pragma unroll
    for (int j = 0; j < 4; ++j) {
      FragBF bf;
      bf.h[0] = *(const u16x8*)(&Bs[(wn + j * 16 + fr) * LDSTR + kbB]);
      bf.h[1] = *(const u16x8*)(&Bs[(wn + j * 16 + fr) * LDSTR + kbB + 16]);
      acc[0][j] = __builtin_amdgcn_wmma_f32_16x16x32_bf16(false, a0.b, false, bf.b, (short)0, acc[0][j], false, false);
      acc[1][j] = __builtin_amdgcn_wmma_f32_16x16x32_bf16(false, a1.b, false, bf.b, (short)0, acc[1][j], false, false);
    }
    __syncthreads();
  }

  // ---- epilogue: alpha, row-scale, bias, store ----
  int fr = lane & 15;
  int rhalf = (lane >> 4) << 3;
  #pragma unroll
  for (int im = 0; im < 2; ++im) {
    #pragma unroll
    for (int jn = 0; jn < 4; ++jn) {
      int col = n0 + wn + jn * 16 + fr;
      float bv = BIAS ? bias[col] : 0.f;
      #pragma unroll
      for (int r = 0; r < 8; ++r) {
        int row = m0 + wm + im * 16 + rhalf + r;
        float v = acc[im][jn][r] * alpha;
        if (RS)   v *= rs[row];
        if (BIAS) v += bv;
        if (OUTF32) ((float*)Cbase)[offC + (long)row * ldc + col] = v;
        else ((unsigned short*)Cbase)[offC + (long)row * ldc + col] = f2bf(v);
      }
    }
  }
}

// ---------------- linear-attention helpers ----------------
__global__ __launch_bounds__(256) void maskkv_kernel(
    const unsigned short* __restrict__ QKV, const int* __restrict__ counts,
    unsigned short* __restrict__ KM, unsigned short* __restrict__ VM, int kk)
{
  int le = blockIdx.y, p = blockIdx.x; int e = le * 4;
  bool valid = p < counts[kk * NEXP + e];
  const unsigned short* q = QKV + ((long)e * CAP + p) * (3 * E_DIM);
  long o = ((long)le * CAP + p) * E_DIM;
  for (int j = threadIdx.x; j < E_DIM; j += 256) {
    KM[o + j] = valid ? q[E_DIM + j] : 0;
    VM[o + j] = valid ? q[2 * E_DIM + j] : 0;
  }
}

__global__ __launch_bounds__(256) void ksum_kernel(
    const unsigned short* __restrict__ KM, float* __restrict__ ksum)
{
  int le = blockIdx.y;
  int j = blockIdx.x * 256 + threadIdx.x;
  const unsigned short* base = KM + (long)le * CAP * E_DIM + j;
  float s = 0.f;
  for (int t = 0; t < CAP; ++t) s += bf2f(base[(long)t * E_DIM]);
  ksum[le * E_DIM + j] = s;
}

__global__ __launch_bounds__(256) void zrow_kernel(
    const unsigned short* __restrict__ QKV, const float* __restrict__ ksum,
    float* __restrict__ z)
{
  int le = blockIdx.y, p = blockIdx.x; int e = le * 4;
  const unsigned short* q = QKV + ((long)e * CAP + p) * (3 * E_DIM);
  const float* ks = ksum + le * E_DIM;
  __shared__ float red[256];
  float s = 0.f;
  for (int j = threadIdx.x; j < E_DIM; j += 256) s += bf2f(q[j]) * ks[j];
  red[threadIdx.x] = s; __syncthreads();
  for (int o = 128; o; o >>= 1) {
    if (threadIdx.x < o) red[threadIdx.x] += red[threadIdx.x + o];
    __syncthreads();
  }
  if (threadIdx.x == 0) z[le * CAP + p] = 1.0f / (red[0] + 1e-6f);
}

// ---------------- masked softmax over attention logits ----------------
__global__ __launch_bounds__(256) void softmax_kernel(
    unsigned short* __restrict__ P, const int* __restrict__ counts, int kk)
{
  int zi = blockIdx.y; int i = zi >> 2; int e = (i / 3) * 4 + (i % 3) + 1;
  int row = blockIdx.x;
  int cnt = counts[kk * NEXP + e];
  unsigned short* p = P + ((long)zi * CAP + row) * CAP;
  int c0 = threadIdx.x, c1 = threadIdx.x + 256;
  float v0 = (c0 < cnt) ? bf2f(p[c0]) : -1e9f;
  float v1 = (c1 < cnt) ? bf2f(p[c1]) : -1e9f;
  __shared__ float red[256];
  red[threadIdx.x] = fmaxf(v0, v1); __syncthreads();
  for (int o = 128; o; o >>= 1) {
    if (threadIdx.x < o) red[threadIdx.x] = fmaxf(red[threadIdx.x], red[threadIdx.x + o]);
    __syncthreads();
  }
  float m = red[0]; __syncthreads();
  float e0 = __expf(v0 - m), e1 = __expf(v1 - m);
  red[threadIdx.x] = e0 + e1; __syncthreads();
  for (int o = 128; o; o >>= 1) {
    if (threadIdx.x < o) red[threadIdx.x] += red[threadIdx.x + o];
    __syncthreads();
  }
  float inv = 1.0f / red[0];
  p[c0] = f2bf(e0 * inv);
  p[c1] = f2bf(e1 * inv);
}

// ---------------- SwiGLU gate ----------------
__global__ __launch_bounds__(256) void silu_kernel(
    const unsigned short* __restrict__ HGL, unsigned short* __restrict__ G)
{
  int e = blockIdx.y, row = blockIdx.x;
  const unsigned short* h = HGL + ((long)e * CAP + row) * (2 * E_DIM);
  unsigned short* g = G + ((long)e * CAP + row) * E_DIM;
  for (int j = threadIdx.x; j < E_DIM; j += 256) {
    float u = bf2f(h[j]);
    float gt = bf2f(h[E_DIM + j]);
    float s = gt / (1.0f + __expf(-gt));
    g[j] = f2bf(u * s);
  }
}

// ---------------- weighted scatter-combine ----------------
__global__ __launch_bounds__(256) void combine_kernel(
    const float* __restrict__ Y, const int* __restrict__ topi,
    const float* __restrict__ topw, const int* __restrict__ tpos,
    float* __restrict__ out, int kk)
{
  int t = blockIdx.x;
  int e = topi[t * TOPK + kk];
  int pos = tpos[t * TOPK + kk];
  float w = topw[t * TOPK + kk];
  const float* y = (pos >= 0) ? (Y + ((long)e * CAP + pos) * H_DIM) : nullptr;
  float* o = out + (long)t * H_DIM;
  for (int j = threadIdx.x; j < H_DIM; j += 256) {
    float base = (kk == 0) ? 0.0f : o[j];
    float add = (pos >= 0) ? w * y[j] : 0.0f;
    o[j] = base + add;
  }
}

// ---------------- host orchestration ----------------
extern "C" void kernel_launch(void* const* d_in, const int* in_sizes, int n_in,
                              void* d_out, int out_size, void* d_ws, size_t ws_size,
                              hipStream_t stream) {
  (void)in_sizes; (void)n_in; (void)out_size; (void)ws_size;
  const float* x        = (const float*)d_in[0];
  const float* router_w = (const float*)d_in[1];
  const float* qkv_w    = (const float*)d_in[2];
  const float* qkv_b    = (const float*)d_in[3];
  const float* sw_w     = (const float*)d_in[4];
  const float* sw_b     = (const float*)d_in[5];
  const float* sv_w     = (const float*)d_in[6];
  const float* sv_b     = (const float*)d_in[7];
  const float* proj_w   = (const float*)d_in[8];
  const float* proj_b   = (const float*)d_in[9];
  float* out = (float*)d_out;

  char* ws = (char*)d_ws;
  size_t off = 0;
  auto alloc = [&](size_t bytes) -> void* {
    void* p = ws + off; off += (bytes + 255) & ~(size_t)255; return p;
  };
  int*   topi  = (int*)  alloc((size_t)T_TOK * TOPK * 4);
  float* topw  = (float*)alloc((size_t)T_TOK * TOPK * 4);
  int*   gidx  = (int*)  alloc((size_t)TOPK * NEXP * CAP * 4);
  int*   tpos  = (int*)  alloc((size_t)T_TOK * TOPK * 4);
  int*   cnts  = (int*)  alloc((size_t)TOPK * NEXP * 4);
  unsigned short* Xb   = (unsigned short*)alloc((size_t)NEXP * CAP * H_DIM * 2);
  unsigned short* QKV  = (unsigned short*)alloc((size_t)NEXP * CAP * 3 * E_DIM * 2);
  unsigned short* Aout = (unsigned short*)alloc((size_t)NEXP * CAP * E_DIM * 2);
  unsigned short* HGL  = (unsigned short*)alloc((size_t)NEXP * CAP * 2 * E_DIM * 2);
  unsigned short* G    = (unsigned short*)alloc((size_t)NEXP * CAP * E_DIM * 2);
  unsigned short* Sb   = (unsigned short*)alloc((size_t)NEXP * CAP * E_DIM * 2);
  float*          Y    = (float*)alloc((size_t)NEXP * CAP * H_DIM * 4);
  unsigned short* KM   = (unsigned short*)alloc((size_t)NLIN * CAP * E_DIM * 2);
  unsigned short* VM   = (unsigned short*)alloc((size_t)NLIN * CAP * E_DIM * 2);
  unsigned short* KV   = (unsigned short*)alloc((size_t)NLIN * E_DIM * E_DIM * 2);
  float*          ksum = (float*)alloc((size_t)NLIN * E_DIM * 4);
  float*          zr   = (float*)alloc((size_t)NLIN * CAP * 4);
  unsigned short* P    = (unsigned short*)alloc((size_t)NSTD * NH * CAP * CAP * 2);

  router_kernel<<<T_TOK, 32, 0, stream>>>(x, router_w, topi, topw);
  assign_kernel<<<1, 128, 0, stream>>>(topi, gidx, tpos, cnts);

  for (int kk = 0; kk < TOPK; ++kk) {
    gather_kernel<<<dim3(CAP, NEXP), 256, 0, stream>>>(x, gidx, cnts, Xb, kk);

    // QKV = Xb @ qkv_w^T + qkv_b      [512,3072] per expert
    gemm_wmma<false, true, true, false, true, false>
      <<<dim3(3 * E_DIM / BN, CAP / BM, NEXP), 256, 0, stream>>>(
        Xb, qkv_w, qkv_b, nullptr, QKV,
        CAP, 3 * E_DIM, H_DIM,
        H_DIM, H_DIM, 3 * E_DIM,
        (long)CAP * H_DIM, (long)3 * E_DIM * H_DIM, (long)CAP * 3 * E_DIM, 3 * E_DIM, 0,
        1.0f, 0);

    // ----- linear attention (experts e%4==0) -----
    maskkv_kernel<<<dim3(CAP, NLIN), 256, 0, stream>>>(QKV, cnts, KM, VM, kk);
    ksum_kernel<<<dim3(E_DIM / 256, NLIN), 256, 0, stream>>>(KM, ksum);
    zrow_kernel<<<dim3(CAP, NLIN), 256, 0, stream>>>(QKV, ksum, zr);

    // KV = KM^T @ VM   [1024,1024] per linear expert
    gemm_wmma<true, false, false, false, false, false>
      <<<dim3(E_DIM / BN, E_DIM / BM, NLIN), 256, 0, stream>>>(
        KM, VM, nullptr, nullptr, KV,
        E_DIM, E_DIM, CAP,
        E_DIM, E_DIM, E_DIM,
        (long)CAP * E_DIM, (long)CAP * E_DIM, (long)E_DIM * E_DIM, 0, 0,
        1.0f, 0);

    // lin_out = (Q @ KV) * z[:,None]  -> Aout (expert e = 4*z)
    gemm_wmma<false, false, false, false, false, true>
      <<<dim3(E_DIM / BN, CAP / BM, NLIN), 256, 0, stream>>>(
        QKV, KV, nullptr, zr, Aout,
        CAP, E_DIM, E_DIM,
        3 * E_DIM, E_DIM, E_DIM,
        (long)4 * CAP * 3 * E_DIM, (long)E_DIM * E_DIM, (long)4 * CAP * E_DIM, 0, (long)CAP,
        1.0f, 0);

    // ----- standard attention (24 experts x 4 heads, batched z=96) -----
    // logits = (q_h @ k_h^T) / 16
    gemm_wmma<false, true, false, false, false, false>
      <<<dim3(CAP / BN, CAP / BM, NSTD * NH), 256, 0, stream>>>(
        QKV, QKV + E_DIM, nullptr, nullptr, P,
        CAP, CAP, HD,
        3 * E_DIM, 3 * E_DIM, CAP,
        (long)CAP * 3 * E_DIM, (long)CAP * 3 * E_DIM, (long)CAP * CAP, 0, 0,
        0.0625f, 1);

    softmax_kernel<<<dim3(CAP, NSTD * NH), 256, 0, stream>>>(P, cnts, kk);

    // attn_out = P @ v_h  -> Aout head slice
    gemm_wmma<false, false, false, false, false, false>
      <<<dim3(HD / BN, CAP / BM, NSTD * NH), 256, 0, stream>>>(
        P, QKV + 2 * E_DIM, nullptr, nullptr, Aout,
        CAP, HD, CAP,
        CAP, 3 * E_DIM, E_DIM,
        (long)CAP * CAP, (long)CAP * 3 * E_DIM, (long)CAP * E_DIM, 0, 0,
        1.0f, 2);

    // ----- SwiGLU + projections -----
    gemm_wmma<false, true, true, false, true, false>
      <<<dim3(2 * E_DIM / BN, CAP / BM, NEXP), 256, 0, stream>>>(
        Aout, sw_w, sw_b, nullptr, HGL,
        CAP, 2 * E_DIM, E_DIM,
        E_DIM, E_DIM, 2 * E_DIM,
        (long)CAP * E_DIM, (long)2 * E_DIM * E_DIM, (long)CAP * 2 * E_DIM, 2 * E_DIM, 0,
        1.0f, 0);

    silu_kernel<<<dim3(CAP, NEXP), 256, 0, stream>>>(HGL, G);

    gemm_wmma<false, true, true, false, true, false>
      <<<dim3(E_DIM / BN, CAP / BM, NEXP), 256, 0, stream>>>(
        G, sv_w, sv_b, nullptr, Sb,
        CAP, E_DIM, E_DIM,
        E_DIM, E_DIM, E_DIM,
        (long)CAP * E_DIM, (long)E_DIM * E_DIM, (long)CAP * E_DIM, E_DIM, 0,
        1.0f, 0);

    gemm_wmma<false, true, true, true, true, false>
      <<<dim3(H_DIM / BN, CAP / BM, NEXP), 256, 0, stream>>>(
        Sb, proj_w, proj_b, nullptr, Y,
        CAP, H_DIM, E_DIM,
        E_DIM, E_DIM, H_DIM,
        (long)CAP * E_DIM, (long)H_DIM * E_DIM, (long)CAP * H_DIM, H_DIM, 0,
        1.0f, 0);

    combine_kernel<<<T_TOK, 256, 0, stream>>>(Y, topi, topw, tpos, out, kk);
  }
}
